// DRBNet_single_20701742367264
// MI455X (gfx1250) — matrix operand order repaired and checked by reading, not compiled, not run on hardware
//
#include <hip/hip_runtime.h>

typedef _Float16 f16;
typedef __attribute__((ext_vector_type(8)))  _Float16 h8;
typedef __attribute__((ext_vector_type(16))) _Float16 h16;
typedef __attribute__((ext_vector_type(8)))  float    f8;

#define LEAKY(v) ((v) > 0.f ? (v) : 0.1f * (v))

// ---------------------------------------------------------------------------
// Implicit-GEMM conv, f16 NHWC (channels padded to 32), fp32 accum via WMMA.
// One wave computes a 32(pixel) x 32(out-channel) tile: two A fragments x two
// B fragments feed four v_wmma_f32_16x16x32_f16 issues per 32-deep K chunk
// (2x A reuse and 2x B reuse).
// A fragment layout (16-bit A 16x32, ISA 7.12.2):
//   lanes 0-15 : row M=lane,    e0..7 = K0..7,  e8..15 = K16..23
//   lanes 16-31: row M=lane-16, e0..7 = K8..15, e8..15 = K24..31
// B fragments are pre-packed so each lane reads 32 contiguous bytes.
// Tile decomposition is mask/shift only (all divisors are powers of two).
// ---------------------------------------------------------------------------
__global__ __launch_bounds__(256) void conv_wmma_kernel(
    const f16* __restrict__ in, int ld_in, int Hin, int Win,
    const f16* __restrict__ wpk, const float* __restrict__ bias,
    f16* __restrict__ out, int ld_out, int co_off,
    const f16* __restrict__ res, int ld_res,
    int Hout, int Wout, int cin_pad, int cout,
    int ks, int stride, int act,
    int tiles, int ntm, int lnt, int wtm, int lwt)
{
  const int lane = threadIdx.x & 31;
  const int wid  = blockIdx.x * 8 + (threadIdx.x >> 5);
  if (wid >= tiles) return;
  const int nt2 = wid & ntm;
  int t = wid >> lnt;
  const int tx = t & wtm;
  const int oy = t >> lwt;
  const int pad = (ks - 1) >> 1;
  const int cbt = cin_pad >> 5;           // 32-channel blocks per tap
  const int KC  = ks * ks * cbt;          // K chunks of 32
  const int m   = lane & 15;
  const int hi  = lane >> 4;
  const int ixA = (tx * 32 + m) * stride - pad;
  const int ixB = ixA + 16 * stride;
  const int iy0 = oy * stride - pad;
  const int pstep = 16 * stride * ld_in;  // halves between the two M sub-tiles

  const h16* w0 = (const h16*)wpk + (size_t)(2 * nt2) * KC * 32 + lane;
  const h16* w1 = w0 + (size_t)KC * 32;
  f8 acc00 = {}, acc01 = {}, acc10 = {}, acc11 = {};
  int kc = 0;
  for (int ky = 0; ky < ks; ++ky) {
    const int iy = iy0 + ky;
    const bool yok = (unsigned)iy < (unsigned)Hin;
    for (int kx = 0; kx < ks; ++kx) {
      const bool okA = yok && (unsigned)(ixA + kx) < (unsigned)Win;
      const bool okB = yok && (unsigned)(ixB + kx) < (unsigned)Win;
      const f16* pA = in + ((size_t)iy * Win + (ixA + kx)) * ld_in + (hi << 3);
      const f16* pB = pA + pstep;
      for (int cb = 0; cb < cbt; ++cb, ++kc) {
        h16 a0, a1;
        if (okA) {
          union { h16 v; h8 p2[2]; } u;
          u.p2[0] = *(const h8*)pA;
          u.p2[1] = *(const h8*)(pA + 16);
          a0 = u.v;
        } else {
          for (int e = 0; e < 16; ++e) a0[e] = (f16)0.f;
        }
        if (okB) {
          union { h16 v; h8 p2[2]; } u;
          u.p2[0] = *(const h8*)pB;
          u.p2[1] = *(const h8*)(pB + 16);
          a1 = u.v;
        } else {
          for (int e = 0; e < 16; ++e) a1[e] = (f16)0.f;
        }
        pA += 32; pB += 32;
        const h16 b0 = w0[(size_t)kc * 32];
        const h16 b1 = w1[(size_t)kc * 32];
        __builtin_prefetch(w0 + (size_t)(kc + 1) * 32, 0, 0);
        acc00 = __builtin_amdgcn_wmma_f32_16x16x32_f16(false, a0, false, b0,
                                                       (short)0, acc00, false, false);
        acc01 = __builtin_amdgcn_wmma_f32_16x16x32_f16(false, a0, false, b1,
                                                       (short)0, acc01, false, false);
        acc10 = __builtin_amdgcn_wmma_f32_16x16x32_f16(false, a1, false, b0,
                                                       (short)0, acc10, false, false);
        acc11 = __builtin_amdgcn_wmma_f32_16x16x32_f16(false, a1, false, b1,
                                                       (short)0, acc11, false, false);
      }
    }
  }

  const int n0 = nt2 * 32 + m;            // out channels within padded range
  const int n1 = n0 + 16;
  const float bv0 = (n0 < cout) ? bias[n0] : 0.f;
  const float bv1 = (n1 < cout) ? bias[n1] : 0.f;
  const size_t row = (size_t)oy * Wout;
  for (int r = 0; r < 8; ++r) {
    const int mm = hi ? r + 8 : r;        // D layout: lanes16-31 hold M=r+8
    const size_t pixA = row + tx * 32 + mm;
    const size_t pixB = pixA + 16;
    float v00 = acc00[r] + bv0, v01 = acc01[r] + bv1;
    float v10 = acc10[r] + bv0, v11 = acc11[r] + bv1;
    if (act) { v00 = LEAKY(v00); v01 = LEAKY(v01);
               v10 = LEAKY(v10); v11 = LEAKY(v11); }
    if (res) {
      v00 += (float)res[pixA * ld_res + n0];
      v01 += (float)res[pixA * ld_res + n1];
      v10 += (float)res[pixB * ld_res + n0];
      v11 += (float)res[pixB * ld_res + n1];
    }
    out[pixA * ld_out + co_off + n0] = (f16)v00;
    out[pixA * ld_out + co_off + n1] = (f16)v01;
    out[pixB * ld_out + co_off + n0] = (f16)v10;
    out[pixB * ld_out + co_off + n1] = (f16)v11;
  }
}

// ---------------------------------------------------------------------------
// ConvTranspose2d(k=4,s=2,p=1) as 4-phase 2x2-tap WMMA conv.
// out[oy,ox] sums taps (ky,kx) with ky==oy (mod 2), kx==ox (mod 2):
//   iy=(oy+ky-2)/2, ix=(ox+kx-2)/2. Always leaky + skip residual.
// 16(pixel) x 32(channel) dual-B tile, mask/shift tile decomposition.
// ---------------------------------------------------------------------------
__global__ __launch_bounds__(256) void upconv_wmma_kernel(
    const f16* __restrict__ in, int ld_in, int Hin, int Win,
    const f16* __restrict__ wpk, const float* __restrict__ bias,
    f16* __restrict__ out, int ld_out,
    const f16* __restrict__ res, int ld_res,
    int cin_pad, int cout, int cout_pad,
    int tiles, int ntm, int lnt, int xtm, int lxt)
{
  const int Hout = Hin * 2, Wout = Win * 2;
  const int lane = threadIdx.x & 31;
  const int wid  = blockIdx.x * 8 + (threadIdx.x >> 5);
  if (wid >= tiles) return;
  const int nt2 = wid & ntm;
  int t = wid >> lnt;
  const int pxp = t & 1; t >>= 1;         // x phase
  const int tx = t & xtm;
  const int oy = t >> lxt;
  const int py = oy & 1;                  // y phase
  const int m = lane & 15, hi = lane >> 4;
  const int ox = tx * 32 + pxp + 2 * m;
  const int cbt = cin_pad >> 5;
  const int KC  = 4 * cbt;                // 2x2 taps
  const h16* w0 = (const h16*)wpk +
                  ((size_t)(py * 2 + pxp) * (cout_pad >> 4) + 2 * nt2) * KC * 32 + lane;
  const h16* w1 = w0 + (size_t)KC * 32;
  f8 acc0 = {}, acc1 = {};
  int kc = 0;
  for (int ay = 0; ay < 2; ++ay) {
    const int iy = (oy + (py + 2 * ay) - 2) >> 1;   // parity-exact
    const bool yok = (unsigned)iy < (unsigned)Hin;
    for (int ax = 0; ax < 2; ++ax) {
      const int ix = (ox + (pxp + 2 * ax) - 2) >> 1;
      const bool ok = yok && (unsigned)ix < (unsigned)Win;
      const f16* p = in + ((size_t)iy * Win + ix) * ld_in + (hi << 3);
      for (int cb = 0; cb < cbt; ++cb, ++kc) {
        h16 a;
        if (ok) {
          union { h16 v; h8 p2[2]; } u;
          u.p2[0] = *(const h8*)p;
          u.p2[1] = *(const h8*)(p + 16);
          a = u.v;
        } else {
          for (int e = 0; e < 16; ++e) a[e] = (f16)0.f;
        }
        p += 32;
        const h16 b0 = w0[(size_t)kc * 32];
        const h16 b1 = w1[(size_t)kc * 32];
        acc0 = __builtin_amdgcn_wmma_f32_16x16x32_f16(false, a, false, b0,
                                                      (short)0, acc0, false, false);
        acc1 = __builtin_amdgcn_wmma_f32_16x16x32_f16(false, a, false, b1,
                                                      (short)0, acc1, false, false);
      }
    }
  }
  const int n0 = nt2 * 32 + m;
  const int n1 = n0 + 16;
  const float bv0 = (n0 < cout) ? bias[n0] : 0.f;
  const float bv1 = (n1 < cout) ? bias[n1] : 0.f;
  for (int r = 0; r < 8; ++r) {
    const int mm  = hi ? r + 8 : r;
    const int oxr = tx * 32 + pxp + 2 * mm;
    const size_t pix = (size_t)oy * Wout + oxr;
    float v0 = LEAKY(acc0[r] + bv0) + (float)res[pix * ld_res + n0];
    float v1 = LEAKY(acc1[r] + bv1) + (float)res[pix * ld_res + n1];
    out[pix * ld_out + n0] = (f16)v0;
    out[pix * ld_out + n1] = (f16)v1;
  }
}

// ---------------------------------------------------------------------------
// Weight repack: fp32 OIHW -> f16 fragment order [nt][kc][lane][16].
// K index = (ky*ks+kx)*cin_pad + ci ; per-lane K = kc*32 + (lane<16?0:8) + (e&7) + (e>>3)*16.
// ---------------------------------------------------------------------------
__global__ void repack_conv_w(const float* __restrict__ w, f16* __restrict__ pk,
                              int cin, int cout, int cin_pad, int cout_pad, int ks)
{
  const size_t total = (size_t)cout_pad * ks * ks * cin_pad;
  size_t idx = (size_t)blockIdx.x * 256 + threadIdx.x;
  if (idx >= total) return;
  const int e    = (int)(idx & 15);
  const int lane = (int)((idx >> 4) & 31);
  size_t r = idx >> 9;
  const int KC = (ks * ks * cin_pad) >> 5;
  const int kc = (int)(r % KC);
  const int nt = (int)(r / KC);
  const int n  = nt * 16 + (lane & 15);
  const int K  = kc * 32 + ((lane >> 4) << 3) + (e & 7) + ((e >> 3) << 4);
  const int tap = K / cin_pad;
  const int ci  = K - tap * cin_pad;
  const int ky = tap / ks, kx = tap - ky * ks;
  float v = 0.f;
  if (n < cout && ci < cin)
    v = w[(((size_t)n * cin + ci) * ks + ky) * ks + kx];
  pk[idx] = (f16)v;
}

// Transposed-conv repack: [phase(py,px)][nt][kc][lane][16]; tap a=(ay,ax),
// ky=py+2*ay, kx=px+2*ax.
__global__ void repack_upconv_w(const float* __restrict__ w, f16* __restrict__ pk,
                                int cin, int cout, int cin_pad, int cout_pad)
{
  const size_t KC = (size_t)cin_pad >> 3;           // 4*cin_pad/32
  const size_t total = 4 * (size_t)(cout_pad >> 4) * KC * 512;
  size_t idx = (size_t)blockIdx.x * 256 + threadIdx.x;
  if (idx >= total) return;
  const int e    = (int)(idx & 15);
  const int lane = (int)((idx >> 4) & 31);
  size_t r = idx >> 9;
  const int kc = (int)(r % KC); r /= KC;
  const int NT = cout_pad >> 4;
  const int nt = (int)(r % NT);
  const int phase = (int)(r / NT);
  const int py = phase >> 1, px = phase & 1;
  const int n = nt * 16 + (lane & 15);
  const int K = kc * 32 + ((lane >> 4) << 3) + (e & 7) + ((e >> 3) << 4);
  const int a4 = K / cin_pad;
  const int ci = K - a4 * cin_pad;
  const int ky = py + 2 * (a4 >> 1), kx = px + 2 * (a4 & 1);
  float v = 0.f;
  if (n < cout && ci < cin)
    v = w[(((size_t)n * cin + ci) * 4 + ky) * 4 + kx];
  pk[idx] = (f16)v;
}

// ---------------------------------------------------------------------------
// Elementwise kernels.
// ---------------------------------------------------------------------------
__global__ void nchw_to_nhwc_f16(const float* __restrict__ src,
                                 f16* __restrict__ dst, int HW, int Cp, int cin)
{
  size_t idx = (size_t)blockIdx.x * 256 + threadIdx.x;
  if (idx >= (size_t)HW * Cp) return;
  const int c = (int)(idx % Cp);
  const size_t pix = idx / Cp;
  dst[idx] = (f16)((c < cin) ? src[(size_t)c * HW + pix] : 0.f);
}

__global__ void down8_kernel(const float* __restrict__ src,
                             f16* __restrict__ dst, int Ho, int Wo)
{
  size_t idx = (size_t)blockIdx.x * 256 + threadIdx.x;
  if (idx >= (size_t)Ho * Wo * 32) return;
  const int c = (int)(idx & 31);
  const size_t pix = idx >> 5;
  if (c >= 3) { dst[idx] = (f16)0.f; return; }
  const int x = (int)(pix % Wo), y = (int)(pix / Wo);
  const int Wi = Wo * 8, Hi = Ho * 8;
  const float* p = src + (size_t)c * Hi * Wi;
  float s = 0.f;
  for (int dy = 0; dy < 8; ++dy)
    for (int dx = 0; dx < 8; ++dx)
      s += p[(size_t)(y * 8 + dy) * Wi + x * 8 + dx];
  dst[idx] = (f16)(s * (1.f / 64.f));
}

__global__ void up2_kernel(const f16* __restrict__ src, f16* __restrict__ dst,
                           int Ho, int Wo)   // output dims, C = 32
{
  size_t idx = (size_t)blockIdx.x * 256 + threadIdx.x;
  if (idx >= (size_t)Ho * Wo * 32) return;
  const int c = (int)(idx & 31);
  const size_t pix = idx >> 5;
  const int x = (int)(pix % Wo), y = (int)(pix / Wo);
  dst[idx] = src[((size_t)(y >> 1) * (Wo >> 1) + (x >> 1)) * 32 + c];
}

// est = img + FAC7x7(img, ker) + r  (img/r/est ld=32, ker ld=64, edge pad)
__global__ void fac_est_kernel(const f16* __restrict__ img,
                               const f16* __restrict__ ker,
                               const f16* __restrict__ rr,
                               f16* __restrict__ est, int H, int W)
{
  size_t idx = (size_t)blockIdx.x * 256 + threadIdx.x;
  if (idx >= (size_t)H * W * 32) return;
  const int c = (int)(idx & 31);
  const size_t pix = idx >> 5;
  if (c >= 3) { est[idx] = (f16)0.f; return; }
  const int x = (int)(pix % W), y = (int)(pix / W);
  float s = (float)img[idx] + (float)rr[idx];
  for (int i = 0; i < 7; ++i) {
    int yy = y + i - 3; yy = yy < 0 ? 0 : (yy >= H ? H - 1 : yy);
    for (int j = 0; j < 7; ++j) {
      int xx = x + j - 3; xx = xx < 0 ? 0 : (xx >= W ? W - 1 : xx);
      s += (float)img[((size_t)yy * W + xx) * 32 + c] *
           (float)ker[pix * 64 + i * 7 + j];
    }
  }
  est[idx] = (f16)s;
}

// Final scale: clip to [-1,1], write fp32 NCHW to d_out.
__global__ void fac_final_kernel(const f16* __restrict__ img,
                                 const f16* __restrict__ ker,
                                 const f16* __restrict__ rr,
                                 float* __restrict__ outp, int H, int W)
{
  const size_t HW = (size_t)H * W;
  size_t idx = (size_t)blockIdx.x * 256 + threadIdx.x;
  if (idx >= 3 * HW) return;
  const int c = (int)(idx / HW);
  const size_t pix = idx % HW;
  const int x = (int)(pix % W), y = (int)(pix / W);
  float s = (float)img[pix * 32 + c] + (float)rr[pix * 32 + c];
  for (int i = 0; i < 7; ++i) {
    int yy = y + i - 3; yy = yy < 0 ? 0 : (yy >= H ? H - 1 : yy);
    for (int j = 0; j < 7; ++j) {
      int xx = x + j - 3; xx = xx < 0 ? 0 : (xx >= W ? W - 1 : xx);
      s += (float)img[((size_t)yy * W + xx) * 32 + c] *
           (float)ker[pix * 64 + i * 7 + j];
    }
  }
  s = s < -1.f ? -1.f : (s > 1.f ? 1.f : s);
  outp[idx] = s;
}

// ---------------------------------------------------------------------------
// Layer table (SPECS order) and host orchestration.
// ---------------------------------------------------------------------------
struct Spec { short cin, cout, ks; };
enum {
  i_conv1_1 = 0, i_conv1_2, i_conv1_3,
  i_conv2_1, i_conv2_2, i_conv2_3,
  i_conv3_1, i_conv3_2, i_conv3_3,
  i_conv4_1, i_conv4_2, i_conv4_3,
  i_conv4_4_a,
  i_rb1_1, i_rb1_2, i_rb2_1, i_rb2_2,
  i_conv4_4_b,
  i_upconv3_u,
  i_up3_1_1, i_up3_1_2, i_up3_2_1, i_up3_2_2,
  i_imgd8_1, i_imgd8_2, i_imgd8_3,
  i_uk3_1, i_uk3_2, i_uk3_3,
  i_ur3_1, i_ur3_2, i_ur3_3,
  i_upconv2_u,
  i_up2_1_1, i_up2_1_2, i_up2_2_1, i_up2_2_2,
  i_imgd4_1, i_imgd4_2, i_imgd4_3,
  i_uk2_1, i_uk2_2, i_uk2_3,
  i_ur2_1, i_ur2_2, i_ur2_3,
  i_imgd2_1, i_imgd2_2, i_imgd2_3,
  i_upconv1_u,
  i_up1_1_1, i_up1_1_2, i_up1_2_1, i_up1_2_2,
  i_imgd1_1, i_imgd1_2, i_imgd1_3,
  i_uk1_1, i_uk1_2, i_uk1_3,
  i_ur1_1, i_ur1_2, i_ur1_3,
  i_uk0_1, i_uk0_2, i_uk0_3,
  i_ur0_1, i_ur0_2, i_ur0_3,
  N_SPECS
};
static const Spec SP[N_SPECS] = {
  {3,32,3},{32,32,3},{32,32,3},
  {32,64,3},{64,64,3},{64,64,3},
  {64,128,3},{128,128,3},{128,128,3},
  {128,256,3},{256,256,3},{256,256,3},
  {256,256,3},
  {256,256,3},{256,256,3},{256,256,3},{256,256,3},
  {256,256,3},
  {256,128,4},
  {128,128,3},{128,128,3},{128,128,3},{128,128,3},
  {3,64,3},{64,128,3},{128,256,3},
  {512,256,3},{256,128,3},{128,49,1},
  {512,256,3},{256,64,3},{64,3,1},
  {128,64,4},
  {64,64,3},{64,64,3},{64,64,3},{64,64,3},
  {3,64,3},{64,128,3},{128,128,3},
  {256,128,3},{128,128,3},{128,49,1},
  {256,128,3},{128,64,3},{64,3,1},
  {3,64,3},{64,64,3},{64,64,3},
  {64,32,4},
  {32,32,3},{32,32,3},{32,32,3},{32,32,3},
  {3,64,3},{64,64,3},{64,32,3},
  {128,64,3},{64,64,3},{64,49,1},
  {128,64,3},{64,64,3},{64,3,1},
  {64,64,3},{64,64,3},{64,49,1},
  {64,64,3},{64,64,3},{64,3,1},
};

extern "C" void kernel_launch(void* const* d_in, const int* in_sizes, int n_in,
                              void* d_out, int out_size, void* d_ws, size_t ws_size,
                              hipStream_t stream) {
  (void)in_sizes; (void)n_in; (void)out_size;
  const float* Cin = (const float*)d_in[0];
  auto W_ = [&](int i) { return (const float*)d_in[1 + 2 * i]; };
  auto B_ = [&](int i) { return (const float*)d_in[2 + 2 * i]; };

  size_t off = 0;
  auto carve = [&](size_t halves) -> f16* {
    f16* p = (f16*)((char*)d_ws + off);
    off += (halves * sizeof(f16) + 255) & ~(size_t)255;
    return p;
  };
  auto pad32 = [](int c) { return (c + 31) & ~31; };

  // --- workspace layout: packed weights ---
  f16* wp[N_SPECS];
  for (int i = 0; i < N_SPECS; ++i) {
    const Spec& s = SP[i];
    size_t cip = pad32(s.cin), cop = pad32(s.cout);
    size_t halves = (s.ks == 4) ? 16 * cop * cip
                                : cop * (size_t)s.ks * s.ks * cip;
    wp[i] = carve(halves);
  }

  // --- workspace layout: activations (f16 NHWC, padded C) ---
  const int H1 = 512, W1 = 512, H2 = 256, W2 = 256, H3 = 128, W3 = 128, H4 = 64, W4 = 64;
  const size_t P1 = (size_t)H1 * W1, P2 = (size_t)H2 * W2, P3 = (size_t)H3 * W3, P4 = (size_t)H4 * W4;
  f16 *x0 = carve(P1 * 32), *t1a = carve(P1 * 32), *t1b = carve(P1 * 32), *f1 = carve(P1 * 32);
  f16 *t2a = carve(P2 * 64), *t2b = carve(P2 * 64), *f2 = carve(P2 * 64);
  f16 *t3a = carve(P3 * 128), *t3b = carve(P3 * 128), *f3 = carve(P3 * 128);
  f16 *t4a = carve(P4 * 256), *t4b = carve(P4 * 256), *t4c = carve(P4 * 256), *t4d = carve(P4 * 256);
  f16 *feat8 = carve(P4 * 512);
  f16 *img8 = carve(P4 * 32), *i8a = carve(P4 * 64), *i8b = carve(P4 * 128);
  f16 *u8a = carve(P4 * 256), *u8b = carve(P4 * 128), *u8c = carve(P4 * 64);
  f16 *k8 = carve(P4 * 64), *r8 = carve(P4 * 32), *est8 = carve(P4 * 32);
  f16 *h3 = carve(P3 * 128), *h3b = carve(P3 * 128);
  f16 *feat4 = carve(P3 * 256);
  f16 *e4i = carve(P3 * 32), *i4a = carve(P3 * 64), *i4b = carve(P3 * 128);
  f16 *u4a = carve(P3 * 128), *u4b = carve(P3 * 128), *u4c = carve(P3 * 64);
  f16 *k4 = carve(P3 * 64), *r4 = carve(P3 * 32), *est4 = carve(P3 * 32);
  f16 *h2 = carve(P2 * 64), *h2b = carve(P2 * 64);
  f16 *feat2 = carve(P2 * 128);
  f16 *e2i = carve(P2 * 32), *i2a = carve(P2 * 64), *i2b = carve(P2 * 64);
  f16 *u2a = carve(P2 * 64), *u2b = carve(P2 * 64);
  f16 *k2 = carve(P2 * 64), *r2 = carve(P2 * 32), *est2 = carve(P2 * 32);
  f16 *h1 = carve(P1 * 32), *h1b = carve(P1 * 32);
  f16 *feat1 = carve(P1 * 64);
  f16 *e1i = carve(P1 * 32), *i1a = carve(P1 * 64), *i1b = carve(P1 * 64);
  f16 *u1a = carve(P1 * 64), *u1b = carve(P1 * 64);
  f16 *k1 = carve(P1 * 64), *r1 = carve(P1 * 32);
  if (off > ws_size) return;  // insufficient scratch

  auto ew = [](size_t n) { return dim3((unsigned)((n + 255) / 256)); };

  // --- repack all weights (deterministic each call) ---
  for (int i = 0; i < N_SPECS; ++i) {
    const Spec& s = SP[i];
    int cip = pad32(s.cin), cop = pad32(s.cout);
    if (s.ks == 4) {
      size_t total = (size_t)16 * cop * cip;
      repack_upconv_w<<<ew(total), 256, 0, stream>>>(W_(i), wp[i], s.cin, s.cout, cip, cop);
    } else {
      size_t total = (size_t)cop * s.ks * s.ks * cip;
      repack_conv_w<<<ew(total), 256, 0, stream>>>(W_(i), wp[i], s.cin, s.cout, cip, cop, s.ks);
    }
  }

  auto conv = [&](int wi, const f16* in, int ld_in, int H, int Wd,
                  f16* out, int ld_out, int co_off, int stride, int act,
                  const f16* res, int ld_res) {
    const Spec& s = SP[wi];
    int cip = pad32(s.cin), cop = pad32(s.cout);
    int Ho = H / stride, Wo = Wd / stride;
    int WT32 = Wo / 32, NT2 = cop / 32;
    int tiles = Ho * WT32 * NT2;
    conv_wmma_kernel<<<(unsigned)((tiles + 7) / 8), 256, 0, stream>>>(
        in, ld_in, H, Wd, wp[wi], B_(wi), out, ld_out, co_off, res, ld_res,
        Ho, Wo, cip, s.cout, s.ks, stride, act,
        tiles, NT2 - 1, __builtin_ctz(NT2), WT32 - 1, __builtin_ctz(WT32));
  };
  auto upconv = [&](int wi, const f16* in, int ld_in, int H, int Wd,
                    f16* out, const f16* res, int ld_res) {
    const Spec& s = SP[wi];
    int cip = pad32(s.cin), cop = pad32(s.cout);
    int Hout = 2 * H, XT = (2 * Wd) / 32, NT2 = cop / 32;
    int tiles = Hout * XT * 2 * NT2;
    upconv_wmma_kernel<<<(unsigned)((tiles + 7) / 8), 256, 0, stream>>>(
        in, ld_in, H, Wd, wp[wi], B_(wi), out, cop, res, ld_res, cip, s.cout, cop,
        tiles, NT2 - 1, __builtin_ctz(NT2), XT - 1, __builtin_ctz(XT));
  };

  // ---------------- forward pass ----------------
  nchw_to_nhwc_f16<<<ew(P1 * 32), 256, 0, stream>>>(Cin, x0, (int)P1, 32, 3);

  // encoder
  conv(i_conv1_1, x0, 32, H1, W1, t1a, 32, 0, 1, 1, nullptr, 0);
  conv(i_conv1_2, t1a, 32, H1, W1, t1b, 32, 0, 1, 1, nullptr, 0);
  conv(i_conv1_3, t1b, 32, H1, W1, f1, 32, 0, 1, 1, nullptr, 0);
  conv(i_conv2_1, f1, 32, H1, W1, t2a, 64, 0, 2, 1, nullptr, 0);
  conv(i_conv2_2, t2a, 64, H2, W2, t2b, 64, 0, 1, 1, nullptr, 0);
  conv(i_conv2_3, t2b, 64, H2, W2, f2, 64, 0, 1, 1, nullptr, 0);
  conv(i_conv3_1, f2, 64, H2, W2, t3a, 128, 0, 2, 1, nullptr, 0);
  conv(i_conv3_2, t3a, 128, H3, W3, t3b, 128, 0, 1, 1, nullptr, 0);
  conv(i_conv3_3, t3b, 128, H3, W3, f3, 128, 0, 1, 1, nullptr, 0);
  conv(i_conv4_1, f3, 128, H3, W3, t4a, 256, 0, 2, 1, nullptr, 0);
  conv(i_conv4_2, t4a, 256, H4, W4, t4b, 256, 0, 1, 1, nullptr, 0);
  conv(i_conv4_3, t4b, 256, H4, W4, t4c, 256, 0, 1, 1, nullptr, 0);
  // bottleneck: conv4_4_a -> rb1 -> rb2 -> conv4_4_b (into feat8[0:256))
  conv(i_conv4_4_a, t4c, 256, H4, W4, t4a, 256, 0, 1, 1, nullptr, 0);
  conv(i_rb1_1, t4a, 256, H4, W4, t4b, 256, 0, 1, 1, nullptr, 0);
  conv(i_rb1_2, t4b, 256, H4, W4, t4d, 256, 0, 1, 0, t4a, 256);
  conv(i_rb2_1, t4d, 256, H4, W4, t4b, 256, 0, 1, 1, nullptr, 0);
  conv(i_rb2_2, t4b, 256, H4, W4, t4c, 256, 0, 1, 0, t4d, 256);
  conv(i_conv4_4_b, t4c, 256, H4, W4, feat8, 512, 0, 1, 1, nullptr, 0);

  // scale 1/8
  down8_kernel<<<ew(P4 * 32), 256, 0, stream>>>(Cin, img8, H4, W4);
  conv(i_imgd8_1, img8, 32, H4, W4, i8a, 64, 0, 1, 1, nullptr, 0);
  conv(i_imgd8_2, i8a, 64, H4, W4, i8b, 128, 0, 1, 1, nullptr, 0);
  conv(i_imgd8_3, i8b, 128, H4, W4, feat8, 512, 256, 1, 1, nullptr, 0);
  conv(i_uk3_1, feat8, 512, H4, W4, u8a, 256, 0, 1, 1, nullptr, 0);
  conv(i_uk3_2, u8a, 256, H4, W4, u8b, 128, 0, 1, 1, nullptr, 0);
  conv(i_uk3_3, u8b, 128, H4, W4, k8, 64, 0, 1, 0, nullptr, 0);
  conv(i_ur3_1, feat8, 512, H4, W4, u8a, 256, 0, 1, 1, nullptr, 0);
  conv(i_ur3_2, u8a, 256, H4, W4, u8c, 64, 0, 1, 1, nullptr, 0);
  conv(i_ur3_3, u8c, 64, H4, W4, r8, 32, 0, 1, 1, nullptr, 0);
  fac_est_kernel<<<ew(P4 * 32), 256, 0, stream>>>(img8, k8, r8, est8, H4, W4);

  // scale 1/4
  upconv(i_upconv3_u, feat8, 512, H4, W4, h3, f3, 128);
  conv(i_up3_1_1, h3, 128, H3, W3, t3a, 128, 0, 1, 1, nullptr, 0);
  conv(i_up3_1_2, t3a, 128, H3, W3, h3b, 128, 0, 1, 0, h3, 128);
  conv(i_up3_2_1, h3b, 128, H3, W3, t3a, 128, 0, 1, 1, nullptr, 0);
  conv(i_up3_2_2, t3a, 128, H3, W3, feat4, 256, 0, 1, 0, h3b, 128);
  up2_kernel<<<ew(P3 * 32), 256, 0, stream>>>(est8, e4i, H3, W3);
  conv(i_imgd4_1, e4i, 32, H3, W3, i4a, 64, 0, 1, 1, nullptr, 0);
  conv(i_imgd4_2, i4a, 64, H3, W3, i4b, 128, 0, 1, 1, nullptr, 0);
  conv(i_imgd4_3, i4b, 128, H3, W3, feat4, 256, 128, 1, 1, nullptr, 0);
  conv(i_uk2_1, feat4, 256, H3, W3, u4a, 128, 0, 1, 1, nullptr, 0);
  conv(i_uk2_2, u4a, 128, H3, W3, u4b, 128, 0, 1, 1, nullptr, 0);
  conv(i_uk2_3, u4b, 128, H3, W3, k4, 64, 0, 1, 0, nullptr, 0);
  conv(i_ur2_1, feat4, 256, H3, W3, u4a, 128, 0, 1, 1, nullptr, 0);
  conv(i_ur2_2, u4a, 128, H3, W3, u4c, 64, 0, 1, 1, nullptr, 0);
  conv(i_ur2_3, u4c, 64, H3, W3, r4, 32, 0, 1, 1, nullptr, 0);
  fac_est_kernel<<<ew(P3 * 32), 256, 0, stream>>>(e4i, k4, r4, est4, H3, W3);

  // scale 1/2
  upconv(i_upconv2_u, feat4, 256, H3, W3, h2, f2, 64);
  conv(i_up2_1_1, h2, 64, H2, W2, t2a, 64, 0, 1, 1, nullptr, 0);
  conv(i_up2_1_2, t2a, 64, H2, W2, h2b, 64, 0, 1, 0, h2, 64);
  conv(i_up2_2_1, h2b, 64, H2, W2, t2a, 64, 0, 1, 1, nullptr, 0);
  conv(i_up2_2_2, t2a, 64, H2, W2, feat2, 128, 0, 1, 0, h2b, 64);
  up2_kernel<<<ew(P2 * 32), 256, 0, stream>>>(est4, e2i, H2, W2);
  conv(i_imgd2_1, e2i, 32, H2, W2, i2a, 64, 0, 1, 1, nullptr, 0);
  conv(i_imgd2_2, i2a, 64, H2, W2, i2b, 64, 0, 1, 1, nullptr, 0);
  conv(i_imgd2_3, i2b, 64, H2, W2, feat2, 128, 64, 1, 1, nullptr, 0);
  conv(i_uk1_1, feat2, 128, H2, W2, u2a, 64, 0, 1, 1, nullptr, 0);
  conv(i_uk1_2, u2a, 64, H2, W2, u2b, 64, 0, 1, 1, nullptr, 0);
  conv(i_uk1_3, u2b, 64, H2, W2, k2, 64, 0, 1, 0, nullptr, 0);
  conv(i_ur1_1, feat2, 128, H2, W2, u2a, 64, 0, 1, 1, nullptr, 0);
  conv(i_ur1_2, u2a, 64, H2, W2, u2b, 64, 0, 1, 1, nullptr, 0);
  conv(i_ur1_3, u2b, 64, H2, W2, r2, 32, 0, 1, 1, nullptr, 0);
  fac_est_kernel<<<ew(P2 * 32), 256, 0, stream>>>(e2i, k2, r2, est2, H2, W2);

  // full res
  upconv(i_upconv1_u, feat2, 128, H2, W2, h1, f1, 32);
  conv(i_up1_1_1, h1, 32, H1, W1, t1a, 32, 0, 1, 1, nullptr, 0);
  conv(i_up1_1_2, t1a, 32, H1, W1, h1b, 32, 0, 1, 0, h1, 32);
  conv(i_up1_2_1, h1b, 32, H1, W1, t1a, 32, 0, 1, 1, nullptr, 0);
  conv(i_up1_2_2, t1a, 32, H1, W1, feat1, 64, 0, 1, 0, h1b, 32);
  up2_kernel<<<ew(P1 * 32), 256, 0, stream>>>(est2, e1i, H1, W1);
  conv(i_imgd1_1, e1i, 32, H1, W1, i1a, 64, 0, 1, 1, nullptr, 0);
  conv(i_imgd1_2, i1a, 64, H1, W1, i1b, 64, 0, 1, 1, nullptr, 0);
  conv(i_imgd1_3, i1b, 64, H1, W1, feat1, 64, 32, 1, 1, nullptr, 0);
  conv(i_uk0_1, feat1, 64, H1, W1, u1a, 64, 0, 1, 1, nullptr, 0);
  conv(i_uk0_2, u1a, 64, H1, W1, u1b, 64, 0, 1, 1, nullptr, 0);
  conv(i_uk0_3, u1b, 64, H1, W1, k1, 64, 0, 1, 0, nullptr, 0);
  conv(i_ur0_1, feat1, 64, H1, W1, u1a, 64, 0, 1, 1, nullptr, 0);
  conv(i_ur0_2, u1a, 64, H1, W1, u1b, 64, 0, 1, 1, nullptr, 0);
  conv(i_ur0_3, u1b, 64, H1, W1, r1, 32, 0, 1, 1, nullptr, 0);
  fac_final_kernel<<<ew(3 * P1), 256, 0, stream>>>(e1i, k1, r1, (float*)d_out, H1, W1);
}
